// HunyuanVideoSingleTransformerBlock_35785667510860
// MI455X (gfx1250) — compile-verified, matrix operand order checked by reading, not run on hardware
//
#include <hip/hip_runtime.h>

// ---------------------------------------------------------------------------
// HunyuanVideo single transformer block on gfx1250 (CDNA5, wave32, WMMA).
// All GEMMs + attention matmuls use v_wmma_f32_16x16x32_bf16.
// GEMM A-tiles staged with async global->LDS copies when available.
// ---------------------------------------------------------------------------

#define HID   3072
#define NH    24
#define HD    128
#define MLPD  12288
#define LSEQ  2304
#define N1    21504   // 3*HID + MLP
#define NCAT  15360   // HID + MLP

typedef __attribute__((ext_vector_type(16))) __bf16 v16bf;
typedef __attribute__((ext_vector_type(8)))  float  v8f;
typedef int v4i_vs __attribute__((vector_size(16)));

union FragBF { unsigned int u[8]; v16bf v; };
union AccF   { float f[8];        v8f  v; };

#if __has_builtin(__builtin_amdgcn_global_load_async_to_lds_b128)
#define HAVE_ASYNC_LDS 1
#else
#define HAVE_ASYNC_LDS 0
#endif

__device__ __forceinline__ void wait_asynccnt0() {
#if __has_builtin(__builtin_amdgcn_s_wait_asynccnt)
  __builtin_amdgcn_s_wait_asynccnt(0);
#else
  asm volatile("s_wait_asynccnt 0" ::: "memory");
#endif
}

__device__ __forceinline__ unsigned short f2bf(float f) {
  unsigned int u = __float_as_uint(f);
  unsigned int r = u + 0x7FFFu + ((u >> 16) & 1u);
  return (unsigned short)(r >> 16);
}
__device__ __forceinline__ float bf2f(unsigned short b) {
  return __uint_as_float(((unsigned int)b) << 16);
}
// K index of the low element of VGPR-pair v for half-lane h (ISA 7.12.2,
// 16-bit A 16x32 layout; same packing reused for our B fragments).
__device__ __forceinline__ int kpair(int v, int h) {
  return (v < 4 ? 2 * v : 16 + 2 * (v - 4)) + 8 * h;
}

// ---------------------------------------------------------------------------
// 1) silu(temb)
__global__ __launch_bounds__(256) void silu_kernel(const float* __restrict__ temb,
                                                   float* __restrict__ st) {
  int i = blockIdx.x * 256 + threadIdx.x;
  if (i < HID) { float t = temb[i]; st[i] = t / (1.f + __expf(-t)); }
}

// 2) emb = silu(temb) @ ada_w + ada_b   (1 x 3072) @ (3072 x 9216)
__global__ __launch_bounds__(256) void ada_kernel(const float* __restrict__ st,
                                                  const float* __restrict__ ada_w,
                                                  const float* __restrict__ ada_b,
                                                  float* __restrict__ emb) {
  int j = blockIdx.x * 256 + threadIdx.x;      // 0..9215
  float acc = 0.f;
  for (int i = 0; i < HID; ++i)
    acc += st[i] * ada_w[(size_t)i * (3 * HID) + j];
  emb[j] = acc + ada_b[j];
}

// 3) LayerNorm + adaLN modulation -> bf16 activations
__global__ __launch_bounds__(256) void ln_mod_kernel(const float* __restrict__ hs,
                                                     const float* __restrict__ emb,
                                                     unsigned short* __restrict__ xn) {
  const int row = blockIdx.x;
  const float* x = hs + (size_t)row * HID;
  float s = 0.f, s2 = 0.f;
  for (int i = threadIdx.x; i < HID; i += 256) { float v = x[i]; s += v; s2 += v * v; }
  #pragma unroll
  for (int m = 16; m; m >>= 1) { s += __shfl_xor(s, m, 32); s2 += __shfl_xor(s2, m, 32); }
  __shared__ float redA[8], redB[8];
  __shared__ float mu_s, ri_s;
  int wave = threadIdx.x >> 5, lane = threadIdx.x & 31;
  if (lane == 0) { redA[wave] = s; redB[wave] = s2; }
  __syncthreads();
  if (threadIdx.x == 0) {
    float ts = 0.f, ts2 = 0.f;
    for (int w = 0; w < 8; ++w) { ts += redA[w]; ts2 += redB[w]; }
    float mu = ts / (float)HID;
    mu_s = mu;
    ri_s = rsqrtf(ts2 / (float)HID - mu * mu + 1e-6f);
  }
  __syncthreads();
  float mu = mu_s, ri = ri_s;
  const float* shift = emb;
  const float* scale = emb + HID;
  for (int i = threadIdx.x; i < HID; i += 256) {
    float v = (x[i] - mu) * ri;
    v = v * (1.f + scale[i]) + shift[i];
    xn[(size_t)row * HID + i] = f2bf(v);
  }
}

// ---------------------------------------------------------------------------
// 4) Tiled WMMA GEMM:  C[MxN] = A(bf16, MxK) @ B(f32->bf16, KxN) + bias
//    mode 0: store bf16.  mode 1: out_f32 = hidden + (acc+bias)*gate.
//    Macro tile 64x64, 8 waves (4 in M x 2 in N), 2 accumulators per wave.
//    A tile staged via async global->LDS (ASYNCcnt) when the builtin exists.
#define ASTR 40   // A-tile LDS stride (shorts); 80B rows -> 16B aligned for b128
#define BSTR 34   // B-tile LDS stride (shorts)
__global__ __launch_bounds__(256) void gemm_bf16_wmma(
    const unsigned short* __restrict__ A, const float* __restrict__ B,
    const float* __restrict__ bias,
    unsigned short* __restrict__ out_bf, float* __restrict__ out_f32,
    const float* __restrict__ hidden, const float* __restrict__ gate,
    int M, int N, int K, int mode) {
  __shared__ unsigned short As[64 * ASTR];   // [m][k], k-contiguous
  __shared__ unsigned short Bs[64 * BSTR];   // [n][k], k-contiguous (transposed)
  const int tid   = threadIdx.x;
  const int lane  = tid & 31;
  const int wave  = tid >> 5;
  const int waveM = wave >> 1;               // 0..3
  const int waveN = wave & 1;                // 0..1
  const int mBase = blockIdx.y * 64;
  const int nBase = blockIdx.x * 64;
  const int lm    = lane & 15;
  const int h     = lane >> 4;

  AccF c0, c1;
  #pragma unroll
  for (int i = 0; i < 8; ++i) { c0.f[i] = 0.f; c1.f[i] = 0.f; }

  const int arow = tid >> 2;                 // 0..63
  const int acol = (tid & 3) * 8;            // 0,8,16,24
  const int bk   = tid >> 3;                 // 0..31
  const int bn   = (tid & 7) * 8;            // 0..56

  for (int k0 = 0; k0 < K; k0 += 32) {
    // ---- A tile 64x32 bf16 ----
    const unsigned short* aptr = A + (size_t)(mBase + arow) * K + k0 + acol;
    unsigned short* asl = &As[arow * ASTR + acol];
#if HAVE_ASYNC_LDS
    // direct global->LDS async copy, 16B per thread (ASYNCcnt tracked)
    __builtin_amdgcn_global_load_async_to_lds_b128(
        (__attribute__((address_space(1))) v4i_vs*)aptr,
        (__attribute__((address_space(3))) v4i_vs*)asl, 0, 0);
#else
    {
      uint4 av = *reinterpret_cast<const uint4*>(aptr);
      *reinterpret_cast<unsigned int*>(asl + 0) = av.x;
      *reinterpret_cast<unsigned int*>(asl + 2) = av.y;
      *reinterpret_cast<unsigned int*>(asl + 4) = av.z;
      *reinterpret_cast<unsigned int*>(asl + 6) = av.w;
    }
#endif
    // ---- B tile 32x64 f32 -> bf16, stored transposed [n][k] ----
    const float* bptr = B + (size_t)(k0 + bk) * N + nBase + bn;
    float4 b0 = reinterpret_cast<const float4*>(bptr)[0];
    float4 b1 = reinterpret_cast<const float4*>(bptr)[1];
    float bv[8] = { b0.x, b0.y, b0.z, b0.w, b1.x, b1.y, b1.z, b1.w };
    #pragma unroll
    for (int j = 0; j < 8; ++j) Bs[(bn + j) * BSTR + bk] = f2bf(bv[j]);
    if (k0 + 32 < K) {                       // -> global_prefetch_b8
      __builtin_prefetch(aptr + 32, 0, 1);
      __builtin_prefetch(bptr + (size_t)32 * N, 0, 1);
    }
#if HAVE_ASYNC_LDS
    wait_asynccnt0();
#endif
    __syncthreads();

    FragBF af, bf0, bf1;
    const int mrow = waveM * 16 + lm;
    const int n0   = waveN * 32 + lm;
    #pragma unroll
    for (int v = 0; v < 8; ++v) {
      int kp = kpair(v, h);
      af.u[v]  = *reinterpret_cast<const unsigned int*>(&As[mrow * ASTR + kp]);
      bf0.u[v] = *reinterpret_cast<const unsigned int*>(&Bs[n0 * BSTR + kp]);
      bf1.u[v] = *reinterpret_cast<const unsigned int*>(&Bs[(n0 + 16) * BSTR + kp]);
    }
    c0.v = __builtin_amdgcn_wmma_f32_16x16x32_bf16(false, af.v, false, bf0.v,
                                                   (short)0, c0.v, false, false);
    c1.v = __builtin_amdgcn_wmma_f32_16x16x32_bf16(false, af.v, false, bf1.v,
                                                   (short)0, c1.v, false, false);
    __syncthreads();
  }

  const int col0 = nBase + waveN * 32 + lm;
  #pragma unroll
  for (int i = 0; i < 8; ++i) {
    int m = mBase + waveM * 16 + i + 8 * h;
    size_t i0 = (size_t)m * N + col0;
    size_t i1 = i0 + 16;
    float v0 = c0.f[i] + bias[col0];
    float v1 = c1.f[i] + bias[col0 + 16];
    if (mode == 0) {
      out_bf[i0] = f2bf(v0);
      out_bf[i1] = f2bf(v1);
    } else {
      out_f32[i0] = hidden[i0] + v0 * gate[col0];
      out_f32[i1] = hidden[i1] + v1 * gate[col0 + 16];
    }
  }
}

// ---------------------------------------------------------------------------
// 5) QKV prep: split heads, RMS-norm q/k, RoPE on image tokens, bf16 out
//    Layout out: [head][l][d] for contiguous per-head attention access.
__global__ __launch_bounds__(128) void qkv_prep_kernel(
    const unsigned short* __restrict__ proj,
    const float* __restrict__ cosT, const float* __restrict__ sinT,
    const float* __restrict__ qn_w, const float* __restrict__ kn_w,
    const int* __restrict__ txt_len,
    unsigned short* __restrict__ qb, unsigned short* __restrict__ kb,
    unsigned short* __restrict__ vb) {
  const int l  = blockIdx.x;
  const int hd = blockIdx.y;
  const int d  = threadIdx.x;                // 0..127
  const int img = LSEQ - *txt_len;
  const unsigned short* row = proj + (size_t)l * N1;
  float q = bf2f(row[(0 * NH + hd) * HD + d]);
  float k = bf2f(row[(1 * NH + hd) * HD + d]);
  float v = bf2f(row[(2 * NH + hd) * HD + d]);
  // RMS over 128 dims across 4 waves
  float sq = q * q, sk = k * k;
  #pragma unroll
  for (int m = 16; m; m >>= 1) { sq += __shfl_xor(sq, m, 32); sk += __shfl_xor(sk, m, 32); }
  __shared__ float rq[4], rk[4];
  int wave = threadIdx.x >> 5, lane = threadIdx.x & 31;
  if (lane == 0) { rq[wave] = sq; rk[wave] = sk; }
  __syncthreads();
  sq = rq[0] + rq[1] + rq[2] + rq[3];
  sk = rk[0] + rk[1] + rk[2] + rk[3];
  q = q * rsqrtf(sq / (float)HD + 1e-6f) * qn_w[d];
  k = k * rsqrtf(sk / (float)HD + 1e-6f) * kn_w[d];
  // RoPE: rot[2i] = -x[2i+1], rot[2i+1] = x[2i]; partner lane = d^1
  float qp = __shfl_xor(q, 1, 32);
  float kp = __shfl_xor(k, 1, 32);
  if (l < img) {
    float c = cosT[(size_t)l * HD + d];
    float s = sinT[(size_t)l * HD + d];
    float qr = (d & 1) ? qp : -qp;
    float kr = (d & 1) ? kp : -kp;
    q = q * c + qr * s;
    k = k * c + kr * s;
  }
  size_t o = ((size_t)hd * LSEQ + l) * HD + d;
  qb[o] = f2bf(q);
  kb[o] = f2bf(k);
  vb[o] = f2bf(v);
}

// ---------------------------------------------------------------------------
// 6) GELU(tanh) on MLP half of proj -> concat[:, 3072:]
__global__ __launch_bounds__(256) void gelu_concat_kernel(
    const unsigned short* __restrict__ proj, unsigned short* __restrict__ cat) {
  size_t idx = (size_t)blockIdx.x * 256 + threadIdx.x;   // over LSEQ*MLPD
  size_t row = idx / MLPD, col = idx % MLPD;
  float x = bf2f(proj[row * N1 + 3 * HID + col]);
  float t = tanhf(0.7978845608028654f * (x + 0.044715f * x * x * x));
  cat[row * NCAT + HID + col] = f2bf(0.5f * x * (1.f + t));
}

// ---------------------------------------------------------------------------
// 7) Flash attention, one wave per (head, 16-query tile). All matmuls WMMA.
#define PSTR 34
__global__ __launch_bounds__(32) void attn_kernel(
    const unsigned short* __restrict__ qb, const unsigned short* __restrict__ kb,
    const unsigned short* __restrict__ vb, unsigned short* __restrict__ cat) {
  const int qTile = blockIdx.x;              // 0..143
  const int head  = blockIdx.y;              // 0..23
  const int lane  = threadIdx.x;
  const int lm    = lane & 15;
  const int h     = lane >> 4;
  const int qBase = qTile * 16;
  const size_t headOff = (size_t)head * LSEQ * HD;

  __shared__ unsigned short Ps[16 * PSTR];   // P tile staged for C->A relayout

  // Q fragments for all 4 K-steps over HD=128 (loaded once)
  FragBF qf[4];
  {
    const unsigned short* qrow = qb + headOff + (size_t)(qBase + lm) * HD;
    #pragma unroll
    for (int kc = 0; kc < 4; ++kc)
      #pragma unroll
      for (int v = 0; v < 8; ++v)
        qf[kc].u[v] = *reinterpret_cast<const unsigned int*>(qrow + kpair(v, h) + kc * 32);
  }

  AccF O[8];
  #pragma unroll
  for (int t = 0; t < 8; ++t)
    #pragma unroll
    for (int i = 0; i < 8; ++i) O[t].f[i] = 0.f;
  float mrow[8], lrow[8];
  #pragma unroll
  for (int i = 0; i < 8; ++i) { mrow[i] = -1e30f; lrow[i] = 0.f; }
  const float scl = 0.08838834764831845f;    // 1/sqrt(128)

  for (int kB = 0; kB < LSEQ; kB += 32) {
    AccF S0, S1;
    #pragma unroll
    for (int i = 0; i < 8; ++i) { S0.f[i] = 0.f; S1.f[i] = 0.f; }
    const unsigned short* kr0 = kb + headOff + (size_t)(kB + lm) * HD;
    const unsigned short* kr1 = kr0 + 16 * HD;
    #pragma unroll
    for (int kc = 0; kc < 4; ++kc) {         // S = Q @ K^T over HD
      FragBF kf0, kf1;
      #pragma unroll
      for (int v = 0; v < 8; ++v) {
        int kp = kpair(v, h) + kc * 32;
        kf0.u[v] = *reinterpret_cast<const unsigned int*>(kr0 + kp);
        kf1.u[v] = *reinterpret_cast<const unsigned int*>(kr1 + kp);
      }
      S0.v = __builtin_amdgcn_wmma_f32_16x16x32_bf16(false, qf[kc].v, false, kf0.v,
                                                     (short)0, S0.v, false, false);
      S1.v = __builtin_amdgcn_wmma_f32_16x16x32_bf16(false, qf[kc].v, false, kf1.v,
                                                     (short)0, S1.v, false, false);
    }
    // online softmax (row i+8h lives in one 16-lane half)
    float f[8];
    #pragma unroll
    for (int i = 0; i < 8; ++i) {
      float s0 = S0.f[i] * scl, s1 = S1.f[i] * scl;
      float mx = fmaxf(s0, s1);
      #pragma unroll
      for (int msk = 1; msk < 16; msk <<= 1) mx = fmaxf(mx, __shfl_xor(mx, msk, 32));
      float mnew = fmaxf(mrow[i], mx);
      f[i] = __expf(mrow[i] - mnew);
      mrow[i] = mnew;
      float p0 = __expf(s0 - mnew), p1 = __expf(s1 - mnew);
      S0.f[i] = p0; S1.f[i] = p1;
      float ps = p0 + p1;
      #pragma unroll
      for (int msk = 1; msk < 16; msk <<= 1) ps += __shfl_xor(ps, msk, 32);
      lrow[i] = lrow[i] * f[i] + ps;
    }
    #pragma unroll
    for (int t = 0; t < 8; ++t)
      #pragma unroll
      for (int i = 0; i < 8; ++i) O[t].f[i] *= f[i];
    // C-layout -> A-fragment relayout through LDS (bf16)
    #pragma unroll
    for (int i = 0; i < 8; ++i) {
      Ps[(i + 8 * h) * PSTR + lm]      = f2bf(S0.f[i]);
      Ps[(i + 8 * h) * PSTR + 16 + lm] = f2bf(S1.f[i]);
    }
    __syncthreads();
    FragBF pf;
    #pragma unroll
    for (int v = 0; v < 8; ++v)
      pf.u[v] = *reinterpret_cast<const unsigned int*>(&Ps[lm * PSTR + kpair(v, h)]);
    // O += P @ V  (8 dim-tiles of 16)
    const unsigned short* vbase = vb + headOff + (size_t)kB * HD;
    #pragma unroll
    for (int t = 0; t < 8; ++t) {
      FragBF vf;
      #pragma unroll
      for (int v = 0; v < 8; ++v) {
        int kp = kpair(v, h);
        unsigned int lo = vbase[(size_t)kp * HD + t * 16 + lm];
        unsigned int hi = vbase[(size_t)(kp + 1) * HD + t * 16 + lm];
        vf.u[v] = lo | (hi << 16);
      }
      O[t].v = __builtin_amdgcn_wmma_f32_16x16x32_bf16(false, pf.v, false, vf.v,
                                                       (short)0, O[t].v, false, false);
    }
    __syncthreads();
  }
  // normalize, write into concat[:, head*128 + d] as bf16
  #pragma unroll
  for (int i = 0; i < 8; ++i) {
    float inv = 1.f / lrow[i];
    int row = qBase + i + 8 * h;
    #pragma unroll
    for (int t = 0; t < 8; ++t)
      cat[(size_t)row * NCAT + head * HD + t * 16 + lm] = f2bf(O[t].f[i] * inv);
  }
}

// ---------------------------------------------------------------------------
extern "C" void kernel_launch(void* const* d_in, const int* in_sizes, int n_in,
                              void* d_out, int out_size, void* d_ws, size_t ws_size,
                              hipStream_t stream) {
  const float* hs      = (const float*)d_in[0];
  const float* temb    = (const float*)d_in[1];
  const float* cosT    = (const float*)d_in[2];
  const float* sinT    = (const float*)d_in[3];
  const float* ada_w   = (const float*)d_in[4];
  const float* ada_b   = (const float*)d_in[5];
  const float* lin1_w  = (const float*)d_in[6];
  const float* lin1_b  = (const float*)d_in[7];
  const float* lin2_w  = (const float*)d_in[8];
  const float* lin2_b  = (const float*)d_in[9];
  const float* qn_w    = (const float*)d_in[10];
  const float* kn_w    = (const float*)d_in[11];
  const int*   txt_len = (const int*)d_in[12];
  float* out = (float*)d_out;

  // workspace carve (256B aligned)
  char* p = (char*)d_ws;
  auto take = [&](size_t bytes) {
    char* r = p;
    p += (bytes + 255) & ~(size_t)255;
    return r;
  };
  float*          silu_b = (float*)take((size_t)HID * 4);
  float*          emb    = (float*)take((size_t)3 * HID * 4);
  unsigned short* xn     = (unsigned short*)take((size_t)LSEQ * HID * 2);
  unsigned short* proj   = (unsigned short*)take((size_t)LSEQ * N1 * 2);
  unsigned short* qb     = (unsigned short*)take((size_t)NH * LSEQ * HD * 2);
  unsigned short* kb     = (unsigned short*)take((size_t)NH * LSEQ * HD * 2);
  unsigned short* vb     = (unsigned short*)take((size_t)NH * LSEQ * HD * 2);
  unsigned short* cat    = (unsigned short*)take((size_t)LSEQ * NCAT * 2);
  const float* gate = emb + 2 * HID;

  silu_kernel<<<(HID + 255) / 256, 256, 0, stream>>>(temb, silu_b);
  ada_kernel<<<(3 * HID) / 256, 256, 0, stream>>>(silu_b, ada_w, ada_b, emb);
  ln_mod_kernel<<<LSEQ, 256, 0, stream>>>(hs, emb, xn);

  // GEMM1: (2304x3072) @ (3072x21504) -> proj (bf16)
  gemm_bf16_wmma<<<dim3(N1 / 64, LSEQ / 64), 256, 0, stream>>>(
      xn, lin1_w, lin1_b, proj, nullptr, nullptr, nullptr,
      LSEQ, N1, HID, 0);

  qkv_prep_kernel<<<dim3(LSEQ, NH), 128, 0, stream>>>(
      proj, cosT, sinT, qn_w, kn_w, txt_len, qb, kb, vb);

  gelu_concat_kernel<<<(unsigned)(((size_t)LSEQ * MLPD) / 256), 256, 0, stream>>>(proj, cat);

  attn_kernel<<<dim3(LSEQ / 16, NH), 32, 0, stream>>>(qb, kb, vb, cat);

  // GEMM2: (2304x15360) @ (15360x3072) + bias, fused residual*gate -> out f32
  gemm_bf16_wmma<<<dim3(HID / 64, LSEQ / 64), 256, 0, stream>>>(
      cat, lin2_w, lin2_b, nullptr, out, hs, gate,
      LSEQ, HID, NCAT, 1);
}
